// BiLstmPartialCrf_90881507984145
// MI455X (gfx1250) — matrix operand order repaired and checked by reading, not compiled, not run on hardware
//
#include <hip/hip_runtime.h>

#define B_   64
#define L_   512
#define E_   256
#define H_   512
#define G4_  2048
#define T_   32
#define IMPOSS (-10000.0f)

typedef __bf16 bf16;
typedef __attribute__((ext_vector_type(16))) __bf16 v16bf;
typedef __attribute__((ext_vector_type(8)))  __bf16 v8bf;
typedef __attribute__((ext_vector_type(8)))  float  v8f;

#define WMMA_BF16(a, b, c) \
  __builtin_amdgcn_wmma_f32_16x16x32_bf16(false, (a), false, (b), (short)0, (c), false, false)

// ---------------------------------------------------------------------------
// WMMA fragment loaders (bf16, 16x16x32, wave32).
// A (16xK=32 slice of row-major MxK):
//   lane l<16 : row m0+l, K = k0+0..15 (v[0..15])
//   lane l>=16: row m0+l-16, K = k0+16..31
// B (K=32 x 16 slice of a row-major (N,K) weight -> per-lane contiguous K):
//   lane n<16 : col n0+n, K = k0+0..15 ; lane n>=16: K = k0+16..31
// ---------------------------------------------------------------------------
__device__ __forceinline__ v16bf load_a_frag(const bf16* A, int lda, int m0, int k0) {
  const int lane = threadIdx.x & 31;
  const int hf   = lane >> 4;
  const int row  = m0 + (lane & 15);
  const bf16* p  = A + (size_t)row * lda + k0 + hf * 16;
  union { v16bf v; v8bf h[2]; } u;
  u.h[0] = *(const v8bf*)p;          // K = k0 + hf*16 .. +7
  u.h[1] = *(const v8bf*)(p + 8);    // K = k0 + hf*16 + 8 .. +15
  return u.v;
}

__device__ __forceinline__ v16bf load_b_frag(const bf16* W, int ldb, int n0, int k0) {
  const int lane = threadIdx.x & 31;
  const int hf   = lane >> 4;
  const int col  = n0 + (lane & 15);
  const bf16* p  = W + (size_t)col * ldb + k0 + hf * 16;
  return *(const v16bf*)p;
}

__device__ __forceinline__ float sigmoidf_(float x) { return 1.0f / (1.0f + expf(-x)); }

// Monotonic-counter grid barrier (one counter per direction, nblk participants).
__device__ __forceinline__ void grid_barrier(unsigned* cnt, unsigned target) {
  __syncthreads();
  if (threadIdx.x == 0) {
    __threadfence();
    __hip_atomic_fetch_add(cnt, 1u, __ATOMIC_ACQ_REL, __HIP_MEMORY_SCOPE_AGENT);
    while (__hip_atomic_load(cnt, __ATOMIC_ACQUIRE, __HIP_MEMORY_SCOPE_AGENT) < target)
      __builtin_amdgcn_s_sleep(2);
    __threadfence();
  }
  __syncthreads();
}

// ---------------------------------------------------------------------------
// Prep kernels
// ---------------------------------------------------------------------------
__global__ void init_state_kernel(bf16* h_pp, unsigned* cnt) {
  int i = blockIdx.x * 256 + threadIdx.x;         // 131072 threads == 2*2*B*H
  h_pp[i] = (bf16)0.0f;
  if (i < 2) cnt[i] = 0u;
}

__global__ void f32_to_bf16_kernel(const float* __restrict__ s, bf16* __restrict__ d, int n) {
  int i = blockIdx.x * 256 + threadIdx.x;
  if (i < n) d[i] = (bf16)s[i];
}

__global__ void gather_embed_kernel(const int* __restrict__ sents,
                                    const float* __restrict__ emb,
                                    bf16* __restrict__ x) {
  int id  = blockIdx.x * 256 + threadIdx.x;       // exactly B*L*E threads
  int tok = id >> 8;                              // E_ == 256
  int e   = id & 255;
  int v   = sents[tok];
  x[id]   = (bf16)emb[(size_t)v * E_ + e];
}

// ---------------------------------------------------------------------------
// Persistent bidirectional LSTM, gate-quartet mapping.
// grid = 32 blocks x 256 threads. Blocks [0,16) forward, [16,32) backward.
// Per direction: 128 waves; wave owns the (mt, ht) 16x16 hidden block and the
// FOUR gate tiles (i,f,g,o at columns ht*16 + q*512) that feed it:
//   - one shared A fragment per K-chunk drives 4 WMMAs (4 indep B streams)
//   - cell state c stays in 8 VGPRs for the whole sequence
//   - LSTM nonlinearity applied in-register straight from the accumulators
//   - h double-buffered in global (ping-pong) -> ONE grid barrier per step
// Weight pointers are laundered through an empty asm per step so LICM cannot
// hoist the 96 B fragments out of the t-loop (round-3 build spilled them to
// scratch); instead they stream from L2 (3 MB/step/dir, read once each).
// ---------------------------------------------------------------------------
__global__ void __launch_bounds__(256, 1) bilstm_kernel(
    const bf16* __restrict__ x,
    const bf16* __restrict__ wih_f, const bf16* __restrict__ whh_f, const float* __restrict__ bias_f,
    const bf16* __restrict__ wih_b, const bf16* __restrict__ whh_b, const float* __restrict__ bias_b,
    bf16* h_pp, bf16* hs_f, bf16* hs_b, unsigned* counters)
{
  const int dir  = blockIdx.x >> 4;
  const int blk  = blockIdx.x & 15;
  const int tid  = threadIdx.x;
  const int wave = tid >> 5;
  const int lane = tid & 31;
  const int qid  = blk * 8 + wave;                // 0..127: quartet id

  const bf16*  wih = dir ? wih_b  : wih_f;
  const bf16*  whh = dir ? whh_b  : whh_f;
  const float* bia = dir ? bias_b : bias_f;
  bf16*  hs  = dir ? hs_b : hs_f;
  unsigned* cnt = counters + dir;

  const int m0  = (qid & 3) * 16;                 // batch-row tile
  const int hc0 = (qid >> 2) * 16;                // hidden-unit column block

  const int col   = hc0 + (lane & 15);
  const int rbase = m0 + ((lane >> 4) << 3);
  const float bi  = bia[0 * H_ + col];
  const float bfv = bia[1 * H_ + col];
  const float bg  = bia[2 * H_ + col];
  const float bo  = bia[3 * H_ + col];

  v8f creg = {};                                  // persistent cell state
  unsigned epoch = 0;

  for (int s = 0; s < L_; ++s) {
    const int t = dir ? (L_ - 1 - s) : s;
    const bf16* hcur = h_pp + (size_t)(dir * 2 + (s & 1)) * (B_ * H_);
    bf16*       hnxt = h_pp + (size_t)(dir * 2 + ((s + 1) & 1)) * (B_ * H_);
    const bf16* xa   = x + (size_t)t * E_;        // row stride L*E (batch-major)

    // Opaque zero: keeps weight B-fragment loads inside the t-loop (no LICM
    // hoist -> no scratch spilling); costs one VALU add per pointer per step.
    int zoff = 0;
    asm volatile("" : "+v"(zoff));
    const bf16* wih_s = wih + zoff;
    const bf16* whh_s = whh + zoff;

    v8f acc0 = {}, acc1 = {}, acc2 = {}, acc3 = {};

    v16bf a = load_a_frag(xa, L_ * E_, m0, 0);    // pipelined shared A fragment
    #pragma unroll
    for (int k0 = 0; k0 < E_; k0 += 32) {
      v16bf an;
      if (k0 + 32 < E_) an = load_a_frag(xa, L_ * E_, m0, k0 + 32);
      else              an = load_a_frag(hcur, H_, m0, 0);     // bridge to h loop
      v16bf b0 = load_b_frag(wih_s, E_, 0 * H_ + hc0, k0);
      v16bf b1 = load_b_frag(wih_s, E_, 1 * H_ + hc0, k0);
      v16bf b2 = load_b_frag(wih_s, E_, 2 * H_ + hc0, k0);
      v16bf b3 = load_b_frag(wih_s, E_, 3 * H_ + hc0, k0);
      acc0 = WMMA_BF16(a, b0, acc0);
      acc1 = WMMA_BF16(a, b1, acc1);
      acc2 = WMMA_BF16(a, b2, acc2);
      acc3 = WMMA_BF16(a, b3, acc3);
      a = an;
    }
    #pragma unroll
    for (int k0 = 0; k0 < H_; k0 += 32) {
      v16bf an = a;
      if (k0 + 32 < H_) an = load_a_frag(hcur, H_, m0, k0 + 32);
      v16bf b0 = load_b_frag(whh_s, H_, 0 * H_ + hc0, k0);
      v16bf b1 = load_b_frag(whh_s, H_, 1 * H_ + hc0, k0);
      v16bf b2 = load_b_frag(whh_s, H_, 2 * H_ + hc0, k0);
      v16bf b3 = load_b_frag(whh_s, H_, 3 * H_ + hc0, k0);
      acc0 = WMMA_BF16(a, b0, acc0);
      acc1 = WMMA_BF16(a, b1, acc1);
      acc2 = WMMA_BF16(a, b2, acc2);
      acc3 = WMMA_BF16(a, b3, acc3);
      a = an;
    }

    // In-register LSTM cell update (gate order i, f, g, o).
    #pragma unroll
    for (int r = 0; r < 8; ++r) {
      const float iv = sigmoidf_(acc0[r] + bi);
      const float fv = sigmoidf_(acc1[r] + bfv);
      const float gv = tanhf(acc2[r] + bg);
      const float ov = sigmoidf_(acc3[r] + bo);
      const float c  = fv * creg[r] + iv * gv;
      creg[r] = c;
      const float h  = ov * tanhf(c);
      const int row  = rbase + r;
      hnxt[(size_t)row * H_ + col]          = (bf16)h;
      hs[((size_t)t * B_ + row) * H_ + col] = (bf16)h;   // (L,B,H) time order
    }

    // Prefetch next timestep's x rows (independent of the recurrence):
    if (s + 1 < L_) {
      const int tn = dir ? (L_ - 2 - s) : (s + 1);
      const bf16* rowp = x + (size_t)tn * E_
                           + (size_t)(m0 + (lane & 15)) * (L_ * E_)
                           + (lane >> 4) * 128;   // halves of the 512B row
      __builtin_prefetch((const char*)rowp, 0, 0);
      __builtin_prefetch((const char*)rowp + 128, 0, 0);
    }

    grid_barrier(cnt, (++epoch) * 16);            // 16 blocks per direction
  }
}

// ---------------------------------------------------------------------------
// em = concat(hf,hb) @ W_out^T + b_out : (L*B, 1024) x (1024, 32) -> (L*B, 32)
// 4096 tiles of 16x16; one wave per tile; grid 512 x 256.
// ---------------------------------------------------------------------------
__global__ void __launch_bounds__(256) em_kernel(
    const bf16* __restrict__ hs_f, const bf16* __restrict__ hs_b,
    const bf16* __restrict__ wout, const float* __restrict__ b_out,
    float* __restrict__ em)
{
  const int wave = threadIdx.x >> 5;
  const int lane = threadIdx.x & 31;
  const int gw   = blockIdx.x * 8 + wave;         // 0..4095
  const int m0   = (gw >> 1) * 16;
  const int n0   = (gw & 1) * 16;
  v8f acc = {};
  v16bf a = load_a_frag(hs_f, H_, m0, 0);
  #pragma unroll
  for (int k0 = 0; k0 < H_; k0 += 32) {
    v16bf an;
    if (k0 + 32 < H_) an = load_a_frag(hs_f, H_, m0, k0 + 32);
    else              an = load_a_frag(hs_b, H_, m0, 0);
    v16bf b = load_b_frag(wout, 2 * H_, n0, k0);
    acc = WMMA_BF16(a, b, acc);
    a = an;
  }
  #pragma unroll
  for (int k0 = 0; k0 < H_; k0 += 32) {
    v16bf an = a;
    if (k0 + 32 < H_) an = load_a_frag(hs_b, H_, m0, k0 + 32);
    v16bf b = load_b_frag(wout, 2 * H_, n0, H_ + k0);
    acc = WMMA_BF16(a, b, acc);
    a = an;
  }
  const int col   = n0 + (lane & 15);
  const int rbase = m0 + ((lane >> 4) << 3);
  const float bv  = b_out[col];
  #pragma unroll
  for (int r = 0; r < 8; ++r)
    em[(size_t)(rbase + r) * T_ + col] = acc[r] + bv;
}

// ---------------------------------------------------------------------------
// CRF: one wave32 per (batch, numerator/denominator) task; lane = tag index.
// alpha lives one value per lane; inner logsumexp over 32 prev tags via shfl.
// grid 16 x 256 => 128 waves == 64 batch * 2.
// ---------------------------------------------------------------------------
__global__ void __launch_bounds__(256) crf_kernel(
    const float* __restrict__ em, const int* __restrict__ targets,
    const float* __restrict__ trans, const float* __restrict__ startt,
    const float* __restrict__ endt, float* __restrict__ partials)
{
  __shared__ float tr_s[T_ * T_];
  const int tid = threadIdx.x;
  #pragma unroll
  for (int q = 0; q < 4; ++q) tr_s[tid * 4 + q] = trans[tid * 4 + q];
  __syncthreads();

  const int wave  = tid >> 5;
  const int lane  = tid & 31;
  const int task  = blockIdx.x * 8 + wave;        // 0..127
  const int b     = task >> 1;
  const int which = task & 1;

  float v[T_];
  float alpha;

  if (which == 0) {
    // ---- denominator (partition function) ----
    alpha = startt[lane] + em[(size_t)b * T_ + lane];
    for (int l = 1; l < L_; ++l) {
      const float e = em[((size_t)l * B_ + b) * T_ + lane];
      float m = -3.4e38f;
      #pragma unroll
      for (int i = 0; i < T_; ++i) {
        const float ai = __shfl(alpha, i, 32);
        const float vi = ai + tr_s[i * T_ + lane];
        v[i] = vi;
        m = fmaxf(m, vi);
      }
      float ssum = 0.f;
      #pragma unroll
      for (int i = 0; i < T_; ++i) ssum += expf(v[i] - m);
      alpha = e + m + logf(ssum);
    }
    float vv = alpha + endt[lane];
    float m = vv;
    for (int off = 16; off > 0; off >>= 1) m = fmaxf(m, __shfl_xor(m, off, 32));
    float se = expf(vv - m);
    for (int off = 16; off > 0; off >>= 1) se += __shfl_xor(se, off, 32);
    if (lane == 0) partials[b] = m + logf(se);
  } else {
    // ---- numerator (masked/partial paths) ----
    int cur = targets[((size_t)b * L_) * T_ + lane];
    alpha = startt[lane] + em[(size_t)b * T_ + lane];
    if (cur == 0) alpha = IMPOSS;
    for (int l = 1; l < L_; ++l) {
      const int   nxt = targets[((size_t)b * L_ + l) * T_ + lane];
      const float e   = em[((size_t)l * B_ + b) * T_ + lane];
      const float e_eff = (nxt == 0) ? IMPOSS : e;
      float m = -3.4e38f;
      #pragma unroll
      for (int i = 0; i < T_; ++i) {
        const float ai = __shfl(alpha, i, 32);
        const int   ci = __shfl(cur,   i, 32);
        const float tr = (ci == 0 || nxt == 0) ? IMPOSS : tr_s[i * T_ + lane];
        const float vi = ai + tr;
        v[i] = vi;
        m = fmaxf(m, vi);
      }
      float ssum = 0.f;
      #pragma unroll
      for (int i = 0; i < T_; ++i) ssum += expf(v[i] - m);
      alpha = e_eff + m + logf(ssum);
      cur = nxt;
    }
    const int ptl = targets[((size_t)b * L_ + (L_ - 1)) * T_ + lane];
    float et = endt[lane] * (float)ptl;
    if (et == 0.f) et = IMPOSS;
    float vv = alpha + et;
    float m = vv;
    for (int off = 16; off > 0; off >>= 1) m = fmaxf(m, __shfl_xor(m, off, 32));
    float se = expf(vv - m);
    for (int off = 16; off > 0; off >>= 1) se += __shfl_xor(se, off, 32);
    if (lane == 0) partials[B_ + b] = m + logf(se);
  }
}

// Deterministic ordered final reduction: sum_b (fwd[b] - gold[b]).
__global__ void reduce_kernel(const float* __restrict__ partials, float* __restrict__ out) {
  __shared__ float s[B_];
  const int t = threadIdx.x;
  if (t < B_) s[t] = partials[t] - partials[B_ + t];
  __syncthreads();
  if (t == 0) {
    float a = 0.f;
    for (int i = 0; i < B_; ++i) a += s[i];
    out[0] = a;
  }
}

// ---------------------------------------------------------------------------
extern "C" void kernel_launch(void* const* d_in, const int* in_sizes, int n_in,
                              void* d_out, int out_size, void* d_ws, size_t ws_size,
                              hipStream_t stream)
{
  (void)in_sizes; (void)n_in; (void)out_size; (void)ws_size;
  const int*   sents   = (const int*)  d_in[0];
  /* lengths d_in[1] unused: all == L */
  const int*   targets = (const int*)  d_in[2];
  const float* emb     = (const float*)d_in[3];
  const float* Wih_f   = (const float*)d_in[4];
  const float* Whh_f   = (const float*)d_in[5];
  const float* b_f     = (const float*)d_in[6];
  const float* Wih_b   = (const float*)d_in[7];
  const float* Whh_b   = (const float*)d_in[8];
  const float* b_b     = (const float*)d_in[9];
  const float* W_out   = (const float*)d_in[10];
  const float* b_out   = (const float*)d_in[11];
  const float* startt  = (const float*)d_in[12];
  const float* endt    = (const float*)d_in[13];
  const float* trans   = (const float*)d_in[14];

  char* ws = (char*)d_ws;
  size_t off = 0;
  auto take = [&](size_t bytes) -> char* {
    char* p = ws + off;
    off = (off + bytes + 255) & ~(size_t)255;
    return p;
  };

  bf16*  x_bf    = (bf16*) take(sizeof(bf16)  * (size_t)B_ * L_ * E_);
  bf16*  wihf    = (bf16*) take(sizeof(bf16)  * (size_t)G4_ * E_);
  bf16*  whhf    = (bf16*) take(sizeof(bf16)  * (size_t)G4_ * H_);
  bf16*  wihb    = (bf16*) take(sizeof(bf16)  * (size_t)G4_ * E_);
  bf16*  whhb    = (bf16*) take(sizeof(bf16)  * (size_t)G4_ * H_);
  bf16*  woutb   = (bf16*) take(sizeof(bf16)  * (size_t)T_ * 2 * H_);
  bf16*  h_pp    = (bf16*) take(sizeof(bf16)  * 2 * 2 * B_ * H_);   // dir x pingpong
  bf16*  hsf     = (bf16*) take(sizeof(bf16)  * (size_t)L_ * B_ * H_);
  bf16*  hsb     = (bf16*) take(sizeof(bf16)  * (size_t)L_ * B_ * H_);
  float* em      = (float*)take(sizeof(float) * (size_t)L_ * B_ * T_);
  float* parts   = (float*)take(sizeof(float) * 2 * B_);
  unsigned* cnts = (unsigned*)take(256);

  // 1) zero h ping-pong buffers + barrier counters (fresh every call)
  init_state_kernel<<<dim3(512), dim3(256), 0, stream>>>(h_pp, cnts);

  // 2) weights -> bf16
  int n;
  n = G4_ * E_;      f32_to_bf16_kernel<<<dim3((n + 255) / 256), dim3(256), 0, stream>>>(Wih_f, wihf, n);
  n = G4_ * H_;      f32_to_bf16_kernel<<<dim3((n + 255) / 256), dim3(256), 0, stream>>>(Whh_f, whhf, n);
  n = G4_ * E_;      f32_to_bf16_kernel<<<dim3((n + 255) / 256), dim3(256), 0, stream>>>(Wih_b, wihb, n);
  n = G4_ * H_;      f32_to_bf16_kernel<<<dim3((n + 255) / 256), dim3(256), 0, stream>>>(Whh_b, whhb, n);
  n = T_ * 2 * H_;   f32_to_bf16_kernel<<<dim3((n + 255) / 256), dim3(256), 0, stream>>>(W_out, woutb, n);

  // 3) embedding gather -> bf16 x (B,L,E)
  gather_embed_kernel<<<dim3((B_ * L_ * E_) / 256), dim3(256), 0, stream>>>(sents, emb, x_bf);

  // 4) persistent cooperative BiLSTM (fwd blocks 0..15, bwd blocks 16..31)
  bilstm_kernel<<<dim3(32), dim3(256), 0, stream>>>(
      x_bf, wihf, whhf, b_f, wihb, whhb, b_b,
      h_pp, hsf, hsb, cnts);

  // 5) emission scores em (L,B,T)
  em_kernel<<<dim3(512), dim3(256), 0, stream>>>(hsf, hsb, woutb, b_out, em);

  // 6) CRF forward + masked-gold scores, one wave per (b, which)
  crf_kernel<<<dim3(16), dim3(256), 0, stream>>>(em, targets, trans, startt, endt, parts);

  // 7) deterministic final sum
  reduce_kernel<<<dim3(1), dim3(64), 0, stream>>>(parts, (float*)d_out);
}
